// KDENet_43868795961770
// MI455X (gfx1250) — matrix operand order repaired and checked by reading, not compiled, not run on hardware
//
#include <hip/hip_runtime.h>

// ---------------------------------------------------------------------------
// 3D sparse U-Net forward (dense-grid formulation) for MI455X / gfx1250.
// All channel GEMMs run on the WMMA pipe: v_wmma_f32_16x16x32_bf16, f32 accum.
// Address math strength-reduced: pow2 shifts for spatial dims, magic-mulhi
// for channel divisions (exact in our ranges).
// ---------------------------------------------------------------------------

#define DG   64
#define CEPS 1e-4f
#define MT   128   // voxels per block (8 waves x 16 rows)
#define KT   32    // K-chunk (bf16 WMMA depth)
#define MAXNT 5    // up to 80 output channels per launch

typedef __bf16 bf16_t;
typedef bf16_t v16bf __attribute__((ext_vector_type(16)));
typedef bf16_t v8bf  __attribute__((ext_vector_type(8)));
typedef float  v8f   __attribute__((ext_vector_type(8)));

// q = n / d (exact for n*d-residual small; magic = ceil(2^32/d), 0 => d==1)
__device__ __forceinline__ int divmg(int n, unsigned mg) {
  return mg ? (int)__umulhi((unsigned)n, mg) : n;
}

// ----------------------------- small kernels -------------------------------

__global__ void k_scatter(const int* __restrict__ coords, const float* __restrict__ feats,
                          float* __restrict__ grid, float* __restrict__ mask, int n)
{
  int i = blockIdx.x * blockDim.x + threadIdx.x;
  if (i < n) {
    int z = coords[3*i+0], y = coords[3*i+1], x = coords[3*i+2];
    int idx = (z * DG + y) * DG + x;
    atomicAdd(&grid[idx], feats[i]);
    mask[idx] = 1.0f;
  }
}

__global__ void k_maskdown(const float* __restrict__ mi, float* __restrict__ mo, int dsh)
{
  int i = blockIdx.x * blockDim.x + threadIdx.x;
  int Do = 1 << dsh, msk = Do - 1;
  int Vo = 1 << (3 * dsh);
  if (i < Vo) {
    int x = i & msk, y = (i >> dsh) & msk, z = i >> (2 * dsh);
    int ish = dsh + 1;
    float m = 0.f;
    for (int dz = 0; dz < 2; ++dz)
      for (int dy = 0; dy < 2; ++dy)
        for (int dx = 0; dx < 2; ++dx)
          m = fmaxf(m, mi[((((2*z+dz) << ish) + 2*y+dy) << ish) + 2*x+dx]);
    mo[i] = m;
  }
}

// Masked per-channel sum / sumsq (block c < C) and mask count (block C).
__global__ void k_bnstats(const float* __restrict__ X, const float* __restrict__ M,
                          float* __restrict__ stats, int V, int C)
{
  __shared__ float s1[256], s2[256];
  int c = blockIdx.x, tid = threadIdx.x;
  float a = 0.f, b = 0.f;
  if (c < C) {
    for (int v = tid; v < V; v += 256) {
      float m = M[v];
      float x = X[(long)v * C + c];
      a += x * m;
      b += x * x * m;   // mask is binary -> x^2*m
    }
  } else {
    for (int v = tid; v < V; v += 256) a += M[v];
  }
  s1[tid] = a; s2[tid] = b;
  __syncthreads();
  for (int s = 128; s > 0; s >>= 1) {
    if (tid < s) { s1[tid] += s1[tid + s]; s2[tid] += s2[tid + s]; }
    __syncthreads();
  }
  if (tid == 0) {
    if (c < C) { stats[c] = s1[0]; stats[256 + c] = s2[0]; }
    else stats[512] = s1[0];
  }
}

__global__ void k_bnrelu(const float* __restrict__ X, const float* __restrict__ M,
                         const float* __restrict__ stats,
                         const float* __restrict__ g, const float* __restrict__ b,
                         float* __restrict__ Y, int V, int C, unsigned cMg)
{
  int tot = V * C;
  int stride = gridDim.x * blockDim.x;
  float n = fmaxf(stats[512], 1.0f);
  for (int i = blockIdx.x * blockDim.x + threadIdx.x; i < tot; i += stride) {
    int v = divmg(i, cMg);
    int c = i - v * C;
    float mu  = stats[c] / n;
    float var = stats[256 + c] / n - mu * mu;
    float y = (X[i] - mu) * rsqrtf(var + CEPS) * g[c] + b[c];
    Y[i] = fmaxf(y, 0.f) * M[v];
  }
}

__global__ void k_copy_cat(const float* __restrict__ S, float* __restrict__ D,
                           int V, int Cs, int Cd, int off, unsigned csMg)
{
  int tot = V * Cs;
  int stride = gridDim.x * blockDim.x;
  for (int i = blockIdx.x * blockDim.x + threadIdx.x; i < tot; i += stride) {
    int v = divmg(i, csMg);
    int c = i - v * Cs;
    D[(long)v * Cd + off + c] = S[i];
  }
}

// ------------------------- implicit-GEMM conv (WMMA) -----------------------
// mode 0: 3x3x3 stride-1 SAME        (K = 27*Cin, Din==Dout)
// mode 1: 2x2x2 stride-2 VALID       (K =  8*Cin, Din==2*Dout)
// mode 2: 2x2x2 transpose, one tap   (K =    Cin, M = coarse voxels, out fine)
// mode 3: 1x1x1                       (K =    Cin)
// Spatial dims passed as log2 (all are powers of two).
// Y[vox*outCS + outCO + n]  (= or += when accum), optionally * Mout[vox].

__global__ __launch_bounds__(256)
void k_conv_wmma(const float* __restrict__ X, const float* __restrict__ W,
                 float* __restrict__ Y, const float* __restrict__ Mout,
                 int shIn, int shOut, int Cin, int Cout,
                 int mode, int tap, int accum, int outCS, int outCO, unsigned cinMg)
{
  __shared__ bf16_t As[MT * KT];            // [row][k]
  __shared__ bf16_t Bs[MAXNT * 16 * KT];    // [n][k] (transposed for frag loads)

  int tid  = threadIdx.x;
  int lane = tid & 31;
  int wv   = tid >> 5;
  int Din  = 1 << shIn;
  int M    = (mode == 2) ? (1 << (3 * shIn)) : (1 << (3 * shOut));
  int T    = (mode == 0) ? 27 : (mode == 1) ? 8 : 1;
  int Ktot = T * Cin;
  int Npad = ((Cout + 15) >> 4) << 4;
  int nNT  = Npad >> 4;
  int base = blockIdx.x * MT;
  int omsk = (1 << shOut) - 1;
  int imsk = Din - 1;

  v8f acc[MAXNT];
#pragma unroll
  for (int i = 0; i < MAXNT; ++i)
#pragma unroll
    for (int j = 0; j < 8; ++j) acc[i][j] = 0.f;

  for (int k0 = 0; k0 < Ktot; k0 += KT) {
    if (k0 + KT < Ktot)  // stream next weight chunk through L2 (global_prefetch_b8)
      __builtin_prefetch(&W[((long)divmg(k0 + KT, cinMg) * Cin) * Cout], 0, 1);
    __syncthreads();

    // ---- stage A tile: MT x KT, f32 -> bf16, implicit-GEMM gather
#pragma unroll
    for (int e = 0; e < (MT * KT) / 256; ++e) {
      int id  = tid + e * 256;
      int row = id >> 5;
      int kk  = id & 31;
      int k   = k0 + kk;
      int vox = base + row;
      float v = 0.f;
      if (k < Ktot && vox < M) {
        int t = divmg(k, cinMg);        // k / Cin  (exact magic-mulhi)
        int c = k - t * Cin;
        long idx = -1;
        if (mode == 0) {
          int ox = vox & omsk, oy = (vox >> shOut) & omsk, oz = vox >> (2 * shOut);
          int tz = (t * 456) >> 12;     // t / 9   (exact, t < 27)
          int tr = t - tz * 9;
          int ty = (tr * 342) >> 10;    // tr / 3  (exact, tr < 9)
          int tx = tr - ty * 3;
          int ix = ox + tx - 1, iy = oy + ty - 1, iz = oz + tz - 1;
          if (ix >= 0 && ix < Din && iy >= 0 && iy < Din && iz >= 0 && iz < Din)
            idx = ((long)((((iz << shIn) + iy) << shIn) + ix) << 0) * Cin + c;
        } else if (mode == 1) {
          int ox = vox & omsk, oy = (vox >> shOut) & omsk, oz = vox >> (2 * shOut);
          int tx = t & 1, ty = (t >> 1) & 1, tz = (t >> 2) & 1;
          idx = ((long)((((2*oz+tz) << shIn) + (2*oy+ty)) << shIn) + (2*ox+tx)) * Cin + c;
        } else {
          idx = (long)vox * Cin + c;
        }
        if (idx >= 0) v = X[idx];
      }
      As[row * KT + kk] = (bf16_t)v;
    }

    // ---- stage B tile transposed: Bs[n][k], f32 -> bf16
    int nb = Npad * KT;
    for (int id = tid; id < nb; id += 256) {
      int n  = id >> 5;
      int kk = id & 31;
      int k  = k0 + kk;
      float v = 0.f;
      if (k < Ktot && n < Cout) {
        int t = divmg(k, cinMg);
        int c = k - t * Cin;
        if (mode == 2) t = tap;
        v = W[((long)t * Cin + c) * Cout + n];
      }
      Bs[n * KT + kk] = (bf16_t)v;
    }
    __syncthreads();

    // ---- fragments (contiguous LDS reads matching gfx1250 WMMA layouts)
    int m    = lane & 15;
    int half = lane >> 4;
    int arow = wv * 16 + m;
    v8bf alo = *(const v8bf*)&As[arow * KT + half * 8];        // K koff..koff+7
    v8bf ahi = *(const v8bf*)&As[arow * KT + half * 8 + 16];   // K koff+16..+23
    v16bf a;
#pragma unroll
    for (int i = 0; i < 8; ++i) { a[i] = alo[i]; a[8 + i] = ahi[i]; }

#pragma unroll
    for (int nt = 0; nt < MAXNT; ++nt) {
      if (nt < nNT) {
        int n = nt * 16 + m;
        v8bf blo = *(const v8bf*)&Bs[n * KT + half * 8];       // K koff..koff+7
        v8bf bhi = *(const v8bf*)&Bs[n * KT + half * 8 + 16];  // K koff+16..+23
        v16bf b;
#pragma unroll
        for (int i = 0; i < 8; ++i) { b[i] = blo[i]; b[8 + i] = bhi[i]; }
        acc[nt] = __builtin_amdgcn_wmma_f32_16x16x32_bf16(
            false, a, false, b, (short)0, acc[nt], false, false);
      }
    }
  }

  // ---- store C/D tile: vgpr r -> M = r + 8*(lane/16), N = lane%16
#pragma unroll
  for (int nt = 0; nt < MAXNT; ++nt) {
    if (nt < nNT) {
#pragma unroll
      for (int r = 0; r < 8; ++r) {
        int mloc = r + 8 * (lane >> 4);
        int vox  = base + wv * 16 + mloc;
        int n    = nt * 16 + (lane & 15);
        if (vox < M && n < Cout) {
          float v = acc[nt][r];
          long ovox;
          if (mode == 2) {
            int ix = vox & imsk, iy = (vox >> shIn) & imsk, iz = vox >> (2 * shIn);
            int tx = tap & 1, ty = (tap >> 1) & 1, tz = (tap >> 2) & 1;
            ovox = (long)((((2*iz+tz) << shOut) + (2*iy+ty)) << shOut) + (2*ix+tx);
          } else {
            ovox = vox;
          }
          if (Mout) v *= Mout[ovox];
          long oi = ovox * (long)outCS + outCO + n;
          if (accum) Y[oi] += v; else Y[oi] = v;
        }
      }
    }
  }
}

// ------------------------------- host side ---------------------------------

static inline const float* FP(void* const* d, int i) { return (const float*)d[i]; }
static inline unsigned mgc(int d) {
  return d <= 1 ? 0u : (unsigned)((0x100000000ULL + (unsigned long long)d - 1) / (unsigned long long)d);
}
static inline int ilog2(int x) { int s = 0; while ((1 << s) < x) ++s; return s; }

// Input pytree leaf indices (JAX sorted-dict flattening):
//  0 coords, 1 features;
//  dec_blocks  base   2: per step st (f=3-st), stride 13:
//    blk0: +0 bn1.b +1 bn1.g +2 bn2.b +3 bn2.g +4 nin +5 w1 +6 w2
//    blk1: +7 bn1.b +8 bn1.g +9 bn2.b +10 bn2.g +11 w1 +12 w2
//  dec_deconv  base  54: per st: b,g,w
//  emb_up      base  66: per st: b,g,w
//  enc_blocks  base  78: level i, block j: 78+12i+6j: bn1.b,bn1.g,bn2.b,bn2.g,w1,w2
//  enc_down    base 138: per i: b,g,w
//  input_conv       150
//  nconvs      base 151: per k: b,g,w      (k=0 -> 80ch ... k=4 -> 16ch)
//  ndensity    base 166: per k: b,g,w,w1

extern "C" void kernel_launch(void* const* d_in, const int* in_sizes, int n_in,
                              void* d_out, int out_size, void* d_ws, size_t ws_size,
                              hipStream_t stream)
{
  (void)n_in; (void)out_size; (void)ws_size;
  const int NP[5] = {16, 32, 48, 64, 80};
  int Dl[5], Vl[5];
  for (int l = 0; l < 5; ++l) { Dl[l] = DG >> l; Vl[l] = Dl[l] * Dl[l] * Dl[l]; }

  char* wsb = (char*)d_ws;
  size_t off = 0;
  auto alloc = [&](size_t elems) -> float* {
    float* p = (float*)(wsb + off);
    off = (off + elems * sizeof(float) + 255) & ~(size_t)255;
    return p;
  };
  float *mask[5], *Xb[5], *SCb[5], *T2b[5], *CATb[4], *ENCb[4];
  for (int l = 0; l < 5; ++l) mask[l] = alloc(Vl[l]);
  float* grid = alloc(Vl[0]);
  for (int l = 0; l < 5; ++l) Xb[l]  = alloc((size_t)Vl[l] * NP[l]);
  for (int l = 0; l < 5; ++l) SCb[l] = alloc((size_t)Vl[l] * NP[l]);
  for (int l = 0; l < 5; ++l) T2b[l] = alloc((size_t)Vl[l] * NP[l]);
  for (int l = 0; l < 4; ++l) CATb[l] = alloc((size_t)Vl[l] * NP[l] * 3);
  for (int l = 0; l < 4; ++l) ENCb[l] = alloc((size_t)Vl[l] * NP[l]);
  float* stats = alloc(513);

  auto conv = [&](const float* X, const float* W, float* Y, const float* Mo,
                  int Din, int Dout, int Cin, int Cout,
                  int mode, int tap, int accum, int oCS, int oCO) {
    int shIn = ilog2(Din), shOut = ilog2(Dout);
    int Mv = (mode == 2) ? Din * Din * Din : Dout * Dout * Dout;
    k_conv_wmma<<<dim3((Mv + MT - 1) / MT), dim3(256), 0, stream>>>(
        X, W, Y, Mo, shIn, shOut, Cin, Cout, mode, tap, accum, oCS, oCO, mgc(Cin));
  };
  auto bnrelu = [&](const float* X, const float* M, const float* gg, const float* bb,
                    float* Y, int V, int C) {
    k_bnstats<<<dim3(C + 1), dim3(256), 0, stream>>>(X, M, stats, V, C);
    long tot = (long)V * C;
    long gb = (tot + 255) / 256; if (gb > 4096) gb = 4096;
    k_bnrelu<<<dim3((int)gb), dim3(256), 0, stream>>>(X, M, stats, gg, bb, Y, V, C, mgc(C));
  };
  auto d2d = [&](float* dst, const float* src, size_t elems) {
    hipMemcpyAsync(dst, src, elems * sizeof(float), hipMemcpyDeviceToDevice, stream);
  };

  // ---- scatter points into dense grid + build mask pyramid
  int npts = in_sizes[1];
  hipMemsetAsync(grid, 0, Vl[0] * sizeof(float), stream);
  hipMemsetAsync(mask[0], 0, Vl[0] * sizeof(float), stream);
  k_scatter<<<dim3((npts + 255) / 256), dim3(256), 0, stream>>>(
      (const int*)d_in[0], (const float*)d_in[1], grid, mask[0], npts);
  for (int l = 1; l < 5; ++l)
    k_maskdown<<<dim3((Vl[l] + 255) / 256), dim3(256), 0, stream>>>(mask[l - 1], mask[l], 6 - l);

  // ---- input conv (1 -> 16)
  conv(grid, FP(d_in, 150), Xb[0], mask[0], 64, 64, 1, 16, 0, 0, 0, 16, 0);
  float* cur = Xb[0];
  float* alt = SCb[0];

  // ---- encoder
  for (int i = 0; i < 5; ++i) {
    int C = NP[i], V = Vl[i], Dd = Dl[i];
    for (int j = 0; j < 2; ++j) {
      int bI = 78 + i * 12 + j * 6;
      d2d(alt, cur, (size_t)V * C);                                    // shortcut
      bnrelu(cur, mask[i], FP(d_in, bI + 1), FP(d_in, bI + 0), cur, V, C);
      conv(cur, FP(d_in, bI + 4), T2b[i], mask[i], Dd, Dd, C, C, 0, 0, 0, C, 0);
      bnrelu(T2b[i], mask[i], FP(d_in, bI + 3), FP(d_in, bI + 2), T2b[i], V, C);
      conv(T2b[i], FP(d_in, bI + 5), alt, mask[i], Dd, Dd, C, C, 0, 0, 1, C, 0);
      float* t = cur; cur = alt; alt = t;
    }
    if (i < 4) {
      d2d(ENCb[i], cur, (size_t)V * C);                                // save skip
      int dI = 138 + 3 * i;
      bnrelu(cur, mask[i], FP(d_in, dI + 1), FP(d_in, dI + 0), cur, V, C);
      conv(cur, FP(d_in, dI + 2), Xb[i + 1], mask[i + 1], Dd, Dd / 2, C, NP[i + 1],
           1, 0, 0, NP[i + 1], 0);
      cur = Xb[i + 1]; alt = SCb[i + 1];
    }
  }

  // ---- decoder (coarse ndensity heads are dead code in the reference; skipped)
  for (int st = 0; st < 4; ++st) {
    int c = 4 - st, f = 3 - st;
    int Cc = NP[c], Cf = NP[f], Vf = Vl[f], Vc = Vl[c], Dc = Dl[c], Df = Dl[f];
    float* EMB = T2b[c];
    int nI = 151 + 3 * st;
    conv(cur, FP(d_in, nI + 2), EMB, mask[c], Dc, Dc, Cc, Cc, 0, 0, 0, Cc, 0);
    bnrelu(EMB, mask[c], FP(d_in, nI + 1), FP(d_in, nI + 0), EMB, Vc, Cc);   // x_emb
    int ddI = 54 + 3 * st;
    bnrelu(cur, mask[c], FP(d_in, ddI + 1), FP(d_in, ddI + 0), cur, Vc, Cc); // x dead after
    for (int t = 0; t < 8; ++t)                                              // x_non -> cat[:,Cf:2Cf]
      conv(cur, FP(d_in, ddI + 2), CATb[f], mask[f], Dc, Df, Cc, Cf, 2, t, 0, 3 * Cf, Cf);
    int euI = 66 + 3 * st;
    bnrelu(EMB, mask[c], FP(d_in, euI + 1), FP(d_in, euI + 0), EMB, Vc, Cc);
    for (int t = 0; t < 8; ++t)                                              // x_emb_up -> cat[:,2Cf:]
      conv(EMB, FP(d_in, euI + 2), CATb[f], mask[f], Dc, Df, Cc, Cf, 2, t, 0, 3 * Cf, 2 * Cf);
    {                                                                        // skip -> cat[:,0:Cf]
      long tot = (long)Vf * Cf;
      long gb = (tot + 255) / 256; if (gb > 4096) gb = 4096;
      k_copy_cat<<<dim3((int)gb), dim3(256), 0, stream>>>(ENCb[f], CATb[f], Vf, Cf, 3 * Cf, 0, mgc(Cf));
    }
    int B = 2 + 13 * st;
    // res block 0: 3Cf -> Cf, with nin shortcut
    conv(CATb[f], FP(d_in, B + 4), SCb[f], nullptr, Df, Df, 3 * Cf, Cf, 3, 0, 0, Cf, 0);
    bnrelu(CATb[f], mask[f], FP(d_in, B + 1), FP(d_in, B + 0), CATb[f], Vf, 3 * Cf);
    conv(CATb[f], FP(d_in, B + 5), T2b[f], mask[f], Df, Df, 3 * Cf, Cf, 0, 0, 0, Cf, 0);
    bnrelu(T2b[f], mask[f], FP(d_in, B + 3), FP(d_in, B + 2), T2b[f], Vf, Cf);
    conv(T2b[f], FP(d_in, B + 6), SCb[f], mask[f], Df, Df, Cf, Cf, 0, 0, 1, Cf, 0);
    cur = SCb[f]; alt = Xb[f];
    // res block 1: Cf -> Cf
    d2d(alt, cur, (size_t)Vf * Cf);
    bnrelu(cur, mask[f], FP(d_in, B + 8), FP(d_in, B + 7), cur, Vf, Cf);
    conv(cur, FP(d_in, B + 11), T2b[f], mask[f], Df, Df, Cf, Cf, 0, 0, 0, Cf, 0);
    bnrelu(T2b[f], mask[f], FP(d_in, B + 10), FP(d_in, B + 9), T2b[f], Vf, Cf);
    conv(T2b[f], FP(d_in, B + 12), alt, mask[f], Df, Df, Cf, Cf, 0, 0, 1, Cf, 0);
    cur = alt; alt = SCb[f];
  }

  // ---- final head at level 0 (nconvs[4]=163..165, ndensity[4]=182..185)
  conv(cur, FP(d_in, 165), T2b[0], mask[0], 64, 64, 16, 16, 0, 0, 0, 16, 0);
  bnrelu(T2b[0], mask[0], FP(d_in, 164), FP(d_in, 163), T2b[0], Vl[0], 16);
  conv(T2b[0], FP(d_in, 184), alt, mask[0], 64, 64, 16, 16, 0, 0, 0, 16, 0);
  bnrelu(alt, mask[0], FP(d_in, 183), FP(d_in, 182), alt, Vl[0], 16);
  conv(alt, FP(d_in, 185), (float*)d_out, mask[0], 64, 64, 16, 2, 3, 0, 0, 2, 0);
}
